// MarioNetteLayerFlax_82729660055745
// MI455X (gfx1250) — matrix operand order repaired and checked by reading, not compiled, not run on hardware
//
#include <hip/hip_runtime.h>
#include <hip/hip_bf16.h>
#include <math.h>

#define N_NODES 20000
#define N_EDGES 160000
#define FEAT 576   // 64*(1+3+5)

typedef __attribute__((ext_vector_type(16))) __bf16 v16bf;
typedef __attribute__((ext_vector_type(8)))  float  v8f;

#define WMMA_BF16(A,B,C) __builtin_amdgcn_wmma_f32_16x16x32_bf16(false,(A),false,(B),(short)0,(C),false,false)

// ---------------- small helpers ----------------

__device__ inline unsigned short f2bf(float f){
  union{float f; unsigned int u;} x; x.f=f;
  unsigned int r = x.u + 0x7fffu + ((x.u>>16)&1u);   // round-to-nearest-even
  return (unsigned short)(r>>16);
}

__device__ inline float fastrcp(float x){ return __builtin_amdgcn_rcpf(x); }

// gelu(tanh approx): 0.5x(1+tanh(g)) == x * sigmoid(2g)  -- branch-free, v_rcp
__device__ inline float geluf(float x){
  float g = 0.7978845608028654f*(x + 0.044715f*x*x*x);
  return x * fastrcp(1.f + __expf(-2.f*g));
}

__device__ inline float susf(float x){ return x>0.f ? __expf(-fastrcp(x)) : 0.f; }

// A-fragment from bf16 row-major LDS activations [16 rows][64 cols].
// ISA 7.12.2 A layout: lane m(0-15),hi=(lane>=16)*8: V0-3 = K kbase+hi..+7 (16B),
// V4-7 = K kbase+hi+16..+23 (16B). Both chunks 16B-aligned -> two ds_load_b128.
__device__ inline v16bf lds_afrag(const unsigned short* actb, int kbase, int lane){
  int m = lane & 15, hi = (lane >> 4) << 3;
  const uint4* p = (const uint4*)(actb + m*64 + kbase + hi);
  union { uint4 q[2]; v16bf bf; } u;
  u.q[0] = p[0];   // K = kbase+hi .. +7
  u.q[1] = p[2];   // K = kbase+hi+16 .. +23
  return u.bf;
}

// A-fragment for K=8 (stage 1, padded to 32): only lanes 0-15, V0-3 carry data.
__device__ inline v16bf lds_afrag8(const unsigned short* actb, int lane){
  union { uint4 q[2]; v16bf bf; } u;
  u.q[0] = make_uint4(0u,0u,0u,0u);
  u.q[1] = make_uint4(0u,0u,0u,0u);
  if (lane < 16) u.q[0] = *(const uint4*)(actb + lane*8);
  return u.bf;
}

// Load pre-packed bf16 B fragment: 256 dwords per 32x16 tile, lane-major, 8 dwords/lane.
__device__ inline v16bf load_bfrag(const uint4* __restrict__ pm, int tile, int lane){
  union { uint4 q[2]; v16bf bf; } u;
  int base = tile*64 + lane*2;
  u.q[0] = pm[base];
  u.q[1] = pm[base+1];
  return u.bf;
}

// Generic small CG contraction: out[k] += w * a[i]*b[j]*cg[i,j,k]
template<int LI,int LF,int LO>
__device__ inline void contract(const float* a, const float* b, const float* cg,
                                float w, float* out){
#pragma unroll
  for (int i=0;i<2*LI+1;++i){
#pragma unroll
    for (int j=0;j<2*LF+1;++j){
      float t = w * a[i]*b[j];
#pragma unroll
      for (int k=0;k<2*LO+1;++k)
        out[k] += t * cg[(i*(2*LF+1)+j)*(2*LO+1)+k];
    }
  }
}

// ---------------- CG coefficient generation (device port of reference) ----------------

struct cdbl { double re, im; };
__device__ inline cdbl cmul(cdbl a, cdbl b){ return {a.re*b.re-a.im*b.im, a.re*b.im+a.im*b.re}; }

__device__ double dfact(int n){ double r=1.0; for(int i=2;i<=n;++i) r*=(double)i; return r; }

__device__ double clebsch(int j1,int m1,int j2,int m2,int j3,int m3){
  if (m1+m2 != m3) return 0.0;
  double pref = sqrt((2.0*j3+1.0)*dfact(j3+j1-j2)*dfact(j3-j1+j2)*dfact(j1+j2-j3)/dfact(j1+j2+j3+1));
  pref *= sqrt(dfact(j3+m3)*dfact(j3-m3)*dfact(j1-m1)*dfact(j1+m1)*dfact(j2-m2)*dfact(j2+m2));
  double s = 0.0;
  for (int k=0;k<=j1+j2-j3;++k){
    int d0=k, d1=j1+j2-j3-k, d2=j1-m1-k, d3=j2+m2-k, d4=j3-j2+m1+k, d5=j3-j1-m2+k;
    if (d0<0||d1<0||d2<0||d3<0||d4<0||d5<0) continue;
    double den = dfact(d0)*dfact(d1)*dfact(d2)*dfact(d3)*dfact(d4)*dfact(d5);
    s += ((k&1)? -1.0 : 1.0)/den;
  }
  return pref*s;
}

__device__ void build_u(int l, cdbl U[5][5]){
  for (int a=0;a<5;++a) for (int b=0;b<5;++b) U[a][b] = {0.0,0.0};
  U[l][l] = {1.0,0.0};
  double r2i = 1.0/sqrt(2.0);
  for (int m=1;m<=l;++m){
    double sgn = (m&1)? -1.0 : 1.0;
    U[l+m][l+m] = {sgn*r2i, 0.0};
    U[l+m][l-m] = {r2i,     0.0};
    U[l-m][l-m] = {0.0,     r2i};
    U[l-m][l+m] = {0.0, -sgn*r2i};
  }
}

__device__ const int PATH_LI[11] = {0,0,0,1,1,1,1,2,2,2,2};
__device__ const int PATH_LF[11] = {0,1,2,0,1,1,2,0,1,2,2};
__device__ const int PATH_LO[11] = {0,1,2,1,0,2,1,2,1,0,2};
__device__ const int CGOFF [11] = {0,1,10,35,44,53,98,143,168,213,238};  // total 363

__global__ void k_cg(float* __restrict__ cgout){
  int p = threadIdx.x;
  if (p >= 11) return;
  int l1=PATH_LI[p], l2=PATH_LF[p], l3=PATH_LO[p];
  int n1=2*l1+1, n2=2*l2+1, n3=2*l3+1;
  double Cc[5][5][5];
  for (int i=0;i<5;++i) for (int j=0;j<5;++j) for (int k=0;k<5;++k) Cc[i][j][k]=0.0;
  for (int m1=-l1;m1<=l1;++m1)
    for (int m2=-l2;m2<=l2;++m2){
      int m3=m1+m2;
      if (m3 < -l3 || m3 > l3) continue;
      Cc[m1+l1][m2+l2][m3+l3] = clebsch(l1,m1,l2,m2,l3,m3);
    }
  cdbl U1[5][5],U2[5][5],U3[5][5];
  build_u(l1,U1); build_u(l2,U2); build_u(l3,U3);
  double Rr[5][5][5], Ri[5][5][5], nr=0.0, ni=0.0;
  for (int a=0;a<n1;++a) for (int b=0;b<n2;++b) for (int c=0;c<n3;++c){
    cdbl s = {0.0,0.0};
    for (int i=0;i<n1;++i) for (int j=0;j<n2;++j) for (int k=0;k<n3;++k){
      double cv = Cc[i][j][k];
      if (cv == 0.0) continue;
      cdbl t = cmul(U1[a][i], U2[b][j]);
      cdbl u3c = {U3[c][k].re, -U3[c][k].im};
      t = cmul(t, u3c);
      s.re += t.re*cv; s.im += t.im*cv;
    }
    Rr[a][b][c]=s.re; Ri[a][b][c]=s.im;
    nr += s.re*s.re; ni += s.im*s.im;
  }
  bool useR = (sqrt(nr) >= sqrt(ni));
  double inv = 1.0/sqrt(useR? nr : ni);
  float* dst = cgout + CGOFF[p];
  for (int a=0;a<n1;++a) for (int b=0;b<n2;++b) for (int c=0;c<n3;++c)
    dst[(a*n2+b)*n3+c] = (float)(((useR)? Rr[a][b][c] : Ri[a][b][c])*inv);
}

// ---------------- pack f32 KxN weight into bf16 WMMA B fragments ----------------
// Tile order: tile = nt*nKt + kt ; per tile 256 dwords: [lane][v] (8 dwords / lane).
// B layout per ISA: VGPR v<4, lane n(0-15): rows {2v,2v+1}, v>=4: rows {16+2(v-4),17+2(v-4)}; lanes 16-31: same +8.
__global__ void k_pack(const float* __restrict__ src, unsigned int* __restrict__ dst,
                       int Ksrc, int N, int nKt, float scale){
  int gid = blockIdx.x*blockDim.x + threadIdx.x;
  int total = nKt*(N/16)*256;
  if (gid >= total) return;
  int t = gid >> 8, r = gid & 255, lane = r >> 3, v = r & 7;
  int kt = t % nKt, nt = t / nKt;
  int n  = nt*16 + (lane & 15);
  int kb = kt*32 + ((lane >> 4) << 3) + 2*v + ((v >= 4) ? 8 : 0);
  float f0 = (kb   < Ksrc) ? src[ kb   *N + n]*scale : 0.f;
  float f1 = (kb+1 < Ksrc) ? src[(kb+1)*N + n]*scale : 0.f;
  dst[gid] = (unsigned int)f2bf(f0) | ((unsigned int)f2bf(f1) << 16);
}

// ---------------- node linears: up & skip ----------------
__global__ void __launch_bounds__(256) k_up_skip(
    const float* __restrict__ nf, const float* __restrict__ Wup,
    const float* __restrict__ Wsk, const int* __restrict__ spec,
    float* __restrict__ up, float* __restrict__ skip){
  __shared__ float xs[4][FEAT];
  int tid = threadIdx.x, n0 = blockIdx.x*4;
  for (int idx=tid; idx<4*FEAT; idx+=256)
    xs[idx/FEAT][idx%FEAT] = nf[(n0+idx/FEAT)*FEAT + (idx%FEAT)];
  __syncthreads();
  int nn = tid>>6, v = tid&63, n = n0+nn;
  int sp = spec[n];
  const float* xb = xs[nn];
  int off = 0;
  for (int l=0;l<3;++l){
    int d = 2*l+1;
    const float* wu = Wup + l*4096;
    const float* ws = Wsk + (sp*3+l)*4096;
    for (int i=0;i<d;++i){
      float au=0.f, as=0.f;
      for (int u=0;u<64;++u){
        float x = xb[off + u*d + i];
        au += x*wu[u*64+v];
        as += x*ws[u*64+v];
      }
      up  [n*FEAT + off + v*d + i] = au*0.125f;   // 1/sqrt(64)
      skip[n*FEAT + off + v*d + i] = as*0.125f;
    }
    off += 64*d;
  }
}

// ---------------- fused edge kernel: radial MLP (WMMA) + SH + CG TP + scatter ----------------
__global__ void __launch_bounds__(128) k_edge(
    const float* __restrict__ vectors, const int* __restrict__ senders,
    const int* __restrict__ receivers, const float* __restrict__ up,
    const float* __restrict__ cgws,
    const uint4* __restrict__ pm1, const uint4* __restrict__ pm2,
    const uint4* __restrict__ pm3, float* __restrict__ agg){
  __shared__ float cgb[363];
  __shared__ float ybuf[16][9];
  __shared__ float emask[16];
  __shared__ __align__(16) unsigned short act0b[16][8];    // bf16 bessel basis
  __shared__ __align__(16) unsigned short act1b[16][64];   // bf16 h1
  __shared__ __align__(16) unsigned short act2b[16][64];   // bf16 h2
  __shared__ float wbuf[16][704];
  int tid = threadIdx.x;
  int e0  = blockIdx.x*16;
  for (int i=tid;i<363;i+=128) cgb[i] = cgws[i];
  if (tid < 16){
    int e = tid;
    float vx=vectors[(e0+e)*3+0], vy=vectors[(e0+e)*3+1], vz=vectors[(e0+e)*3+2];
    float len = sqrtf(vx*vx+vy*vy+vz*vz);
    emask[e] = (len==0.f) ? 0.f : 1.f;
    float il = 1.f/fmaxf(len, 1e-9f);
    float x = vx*il, y = vy*il, z = vz*il;
    ybuf[e][0]=1.f;
    const float s3=1.7320508075688772f;
    ybuf[e][1]=s3*y; ybuf[e][2]=s3*z; ybuf[e][3]=s3*x;
    const float s15=3.872983346207417f, s5h=1.118033988749895f, s15h=1.9364916731037085f;
    ybuf[e][4]=s15*x*y; ybuf[e][5]=s15*y*z; ybuf[e][6]=s5h*(3.f*z*z-1.f);
    ybuf[e][7]=s15*x*z; ybuf[e][8]=s15h*(x*x-y*y);
    float xl = len, safe = (xl==0.f)?1.f:xl, env = 0.f;
    if (xl < 1.f) env = __expf(2.f*xl*xl/fminf(xl*xl-1.f,-1e-6f));
    float benv = 1.4142135623730951f*env*fastrcp(safe);
    for (int nb=0;nb<8;++nb)
      act0b[e][nb] = f2bf(benv*__sinf(3.14159265358979323846f*(float)(nb+1)*xl));
  }
  __syncthreads();
  int wave = tid>>5, lane = tid&31;
  int col  = (lane&15), rbase = (lane>>4)<<3;
  // stage 1: h1 = gelu(r @ M1/sqrt(8))   (K padded 8->32, scale folded into pm1)
  {
    v8f c = {0,0,0,0,0,0,0,0};
    v16bf a = lds_afrag8(&act0b[0][0], lane);
    v16bf b = load_bfrag(pm1, wave, lane);
    c = WMMA_BF16(a,b,c);
#pragma unroll
    for (int j=0;j<8;++j) act1b[rbase+j][wave*16+col] = f2bf(geluf(c[j]));
  }
  __syncthreads();
  // stage 2: h2 = gelu(h1 @ M2/8)
  {
    v8f c = {0,0,0,0,0,0,0,0};
    for (int kt=0;kt<2;++kt){
      v16bf a = lds_afrag(&act1b[0][0], kt*32, lane);
      v16bf b = load_bfrag(pm2, wave*2+kt, lane);
      c = WMMA_BF16(a,b,c);
    }
#pragma unroll
    for (int j=0;j<8;++j) act2b[rbase+j][wave*16+col] = f2bf(geluf(c[j]));
  }
  __syncthreads();
  // stage 3: w = (h2 @ M3/8) masked   (44 column tiles, 11 per wave)
  for (int t=0;t<11;++t){
    int nt = wave + 4*t;
    v8f c = {0,0,0,0,0,0,0,0};
    for (int kt=0;kt<2;++kt){
      v16bf a = lds_afrag(&act2b[0][0], kt*32, lane);
      v16bf b = load_bfrag(pm3, nt*2+kt, lane);
      c = WMMA_BF16(a,b,c);
    }
#pragma unroll
    for (int j=0;j<8;++j) wbuf[rbase+j][nt*16+col] = c[j]*emask[rbase+j];
  }
  __syncthreads();
  // stage 4: per (edge, channel) tensor product + atomic scatter
  for (int task=tid; task<1024; task+=128){
    int e = task>>6, v = task&63, ge = e0+e;
    int s = senders[ge], r = receivers[ge];
    const float* ub = up + s*FEAT;
    float m0[1]; m0[0] = ub[v];
    float m1[3], m2[5];
#pragma unroll
    for (int i=0;i<3;++i) m1[i] = ub[64  + v*3 + i];
#pragma unroll
    for (int i=0;i<5;++i) m2[i] = ub[256 + v*5 + i];
    float y[9];
#pragma unroll
    for (int i=0;i<9;++i) y[i] = ybuf[e][i];
    float o0[1]={0.f}, o1[3]={0.f,0.f,0.f}, o2[5]={0.f,0.f,0.f,0.f,0.f};
    const float* wr = wbuf[e];
    contract<0,0,0>(m0, y+0, cgb+0,   wr[0*64+v], o0);
    contract<0,1,1>(m0, y+1, cgb+1,   wr[1*64+v], o1);
    contract<0,2,2>(m0, y+4, cgb+10,  wr[2*64+v], o2);
    contract<1,0,1>(m1, y+0, cgb+35,  wr[3*64+v], o1);
    contract<1,1,0>(m1, y+1, cgb+44,  wr[4*64+v], o0);
    contract<1,1,2>(m1, y+1, cgb+53,  wr[5*64+v], o2);
    contract<1,2,1>(m1, y+4, cgb+98,  wr[6*64+v], o1);
    contract<2,0,2>(m2, y+0, cgb+143, wr[7*64+v], o2);
    contract<2,1,1>(m2, y+1, cgb+168, wr[8*64+v], o1);
    contract<2,2,0>(m2, y+4, cgb+213, wr[9*64+v], o0);
    contract<2,2,2>(m2, y+4, cgb+238, wr[10*64+v], o2);
    const float sc = 0.3535533905932738f;   // 1/sqrt(AVG_NEIGH=8)
    float* ab = agg + r*FEAT;
    atomicAdd(&ab[v], o0[0]*sc);
#pragma unroll
    for (int i=0;i<3;++i) atomicAdd(&ab[64  + v*3 + i], o1[i]*sc);
#pragma unroll
    for (int i=0;i<5;++i) atomicAdd(&ab[256 + v*5 + i], o2[i]*sc);
  }
}

// ---------------- per-node post: down / squares / W_out / soft-norm / final ----------------
__global__ void __launch_bounds__(256) k_post(
    const float* __restrict__ agg, const float* __restrict__ skip,
    const float* __restrict__ Wdn, const float* __restrict__ Wo0,
    const float* __restrict__ Wo1, const float* __restrict__ Wo2,
    const float* __restrict__ cgws, float* __restrict__ out){
  __shared__ float cgb[363];
  __shared__ float xs[4][FEAT];
  __shared__ float dnb[4][FEAT];
  __shared__ float cat0[4][256];
  __shared__ float cat1[4][192][3];
  __shared__ float cat2[4][256][5];
  int tid = threadIdx.x, n0 = blockIdx.x*4;
  for (int i=tid;i<363;i+=256) cgb[i] = cgws[i];
  for (int idx=tid; idx<4*FEAT; idx+=256)
    xs[idx/FEAT][idx%FEAT] = agg[(n0+idx/FEAT)*FEAT + (idx%FEAT)];
  __syncthreads();
  int nn = tid>>6, v = tid&63, n = n0+nn;
  {
    int off = 0;
    for (int l=0;l<3;++l){
      int d = 2*l+1;
      const float* w = Wdn + l*4096;
      for (int i=0;i<d;++i){
        float acc = 0.f;
        for (int u=0;u<64;++u) acc += xs[nn][off + u*d + i]*w[u*64+v];
        acc *= 0.125f;
        if (l==0) acc = geluf(acc);
        dnb[nn][off + v*d + i] = acc;
      }
      off += 64*d;
    }
  }
  __syncthreads();
  {
    float d0[1]; d0[0] = dnb[nn][v];
    float d1[3], d2[5];
#pragma unroll
    for (int i=0;i<3;++i) d1[i] = dnb[nn][64  + v*3 + i];
#pragma unroll
    for (int i=0;i<5;++i) d2[i] = dnb[nn][256 + v*5 + i];
    float s000[1]={0.f}, s110[1]={0.f}, s220[1]={0.f};
    float s011[3]={0.f,0.f,0.f}, s121[3]={0.f,0.f,0.f};
    float s022[5]={0.f,0.f,0.f,0.f,0.f}, s112[5]={0.f,0.f,0.f,0.f,0.f}, s222[5]={0.f,0.f,0.f,0.f,0.f};
    contract<0,0,0>(d0,d0,cgb+0,   1.f,s000);
    contract<1,1,0>(d1,d1,cgb+44,  1.f,s110);
    contract<2,2,0>(d2,d2,cgb+213, 1.f,s220);
    contract<0,1,1>(d0,d1,cgb+1,   1.f,s011);
    contract<1,2,1>(d1,d2,cgb+98,  1.f,s121);
    contract<0,2,2>(d0,d2,cgb+10,  1.f,s022);
    contract<1,1,2>(d1,d1,cgb+53,  1.f,s112);
    contract<2,2,2>(d2,d2,cgb+238, 1.f,s222);
    cat0[nn][v]=d0[0]; cat0[nn][64+v]=s000[0]; cat0[nn][128+v]=s110[0]; cat0[nn][192+v]=s220[0];
#pragma unroll
    for (int i=0;i<3;++i){ cat1[nn][v][i]=d1[i]; cat1[nn][64+v][i]=s011[i]; cat1[nn][128+v][i]=s121[i]; }
#pragma unroll
    for (int i=0;i<5;++i){ cat2[nn][v][i]=d2[i]; cat2[nn][64+v][i]=s022[i]; cat2[nn][128+v][i]=s112[i]; cat2[nn][192+v][i]=s222[i]; }
  }
  __syncthreads();
  {
    float o0 = 0.f;
    for (int u=0;u<256;++u) o0 += cat0[nn][u]*Wo0[u*64+v];
    o0 *= 0.0625f;                                  // 1/sqrt(256)
    float o1[3]={0.f,0.f,0.f};
    for (int u=0;u<192;++u){
      float w = Wo1[u*64+v];
#pragma unroll
      for (int i=0;i<3;++i) o1[i] += cat1[nn][u][i]*w;
    }
#pragma unroll
    for (int i=0;i<3;++i) o1[i] *= 0.07216878364870323f;  // 1/sqrt(192)
    float o2[5]={0.f,0.f,0.f,0.f,0.f};
    for (int u=0;u<256;++u){
      float w = Wo2[u*64+v];
#pragma unroll
      for (int i=0;i<5;++i) o2[i] += cat2[nn][u][i]*w;
    }
#pragma unroll
    for (int i=0;i<5;++i) o2[i] *= 0.0625f;
    // soft-norm (fast rcp: precision irrelevant after bf16 WMMA path)
    float nrm, iden, b0, b1[3], b2[5];
    nrm = fabsf(o0)*1e-5f; iden = fastrcp(1.f + nrm*susf(nrm)); b0 = o0*iden;
    nrm = sqrtf(o1[0]*o1[0]+o1[1]*o1[1]+o1[2]*o1[2])*1e-5f;
    iden = fastrcp(1.f + nrm*susf(nrm));
#pragma unroll
    for (int i=0;i<3;++i) b1[i] = o1[i]*iden;
    nrm = sqrtf(o2[0]*o2[0]+o2[1]*o2[1]+o2[2]*o2[2]+o2[3]*o2[3]+o2[4]*o2[4])*1e-5f;
    iden = fastrcp(1.f + nrm*susf(nrm));
#pragma unroll
    for (int i=0;i<5;++i) b2[i] = o2[i]*iden;
    const float* sk = skip + n*FEAT;
    float* ob = out + n*FEAT;
    ob[v] = 0.9f*sk[v] + 0.45f*b0;
#pragma unroll
    for (int i=0;i<3;++i) ob[64  + v*3 + i] = 0.9f*sk[64  + v*3 + i] + 0.45f*b1[i];
#pragma unroll
    for (int i=0;i<5;++i) ob[256 + v*5 + i] = 0.9f*sk[256 + v*5 + i] + 0.45f*b2[i];
  }
}

// ---------------- host launch ----------------
extern "C" void kernel_launch(void* const* d_in, const int* in_sizes, int n_in,
                              void* d_out, int out_size, void* d_ws, size_t ws_size,
                              hipStream_t stream) {
  (void)in_sizes; (void)n_in; (void)out_size; (void)ws_size;
  const float* vectors     = (const float*)d_in[0];
  const float* node_feats  = (const float*)d_in[1];
  const float* W_up        = (const float*)d_in[2];
  const float* W_skip      = (const float*)d_in[3];
  const float* M1          = (const float*)d_in[4];
  const float* M2          = (const float*)d_in[5];
  const float* M3          = (const float*)d_in[6];
  const float* W_out0      = (const float*)d_in[7];
  const float* W_out1      = (const float*)d_in[8];
  const float* W_out2      = (const float*)d_in[9];
  const float* W_down      = (const float*)d_in[10];
  const int*   node_specie = (const int*)d_in[11];
  const int*   senders     = (const int*)d_in[12];
  const int*   receivers   = (const int*)d_in[13];
  float* out = (float*)d_out;

  char* ws = (char*)d_ws;
  float*        cg  = (float*)(ws + 0);            // 363 f32
  unsigned int* pm1 = (unsigned int*)(ws + 4096);  // 1024 dwords
  unsigned int* pm2 = (unsigned int*)(ws + 8192);  // 2048 dwords
  unsigned int* pm3 = (unsigned int*)(ws + 16384); // 22528 dwords
  float* up   = (float*)(ws + 131072);
  float* skip = up   + (size_t)N_NODES*FEAT;
  float* agg  = skip + (size_t)N_NODES*FEAT;

  k_cg<<<1, 16, 0, stream>>>(cg);
  k_pack<<<4,  256, 0, stream>>>(M1, pm1, 8,  64,  1, 0.35355339059327f); // 1/sqrt(8)
  k_pack<<<8,  256, 0, stream>>>(M2, pm2, 64, 64,  2, 0.125f);            // 1/sqrt(64)
  k_pack<<<88, 256, 0, stream>>>(M3, pm3, 64, 704, 2, 0.125f);
  hipMemsetAsync(agg, 0, (size_t)N_NODES*FEAT*sizeof(float), stream);
  k_up_skip<<<N_NODES/4, 256, 0, stream>>>(node_feats, W_up, W_skip, node_specie, up, skip);
  k_edge<<<N_EDGES/16, 128, 0, stream>>>(vectors, senders, receivers, up, cg,
                                         (const uint4*)pm1, (const uint4*)pm2,
                                         (const uint4*)pm3, agg);
  k_post<<<N_NODES/4, 256, 0, stream>>>(agg, skip, W_down, W_out0, W_out1, W_out2, cg, out);
}
